// DeepFieldWeightedFactorizationMachineModel_74783970558603
// MI455X (gfx1250) — matrix operand order repaired and checked by reading, not compiled, not run on hardware
//
#include <hip/hip_runtime.h>

typedef _Float16 h16;
typedef __attribute__((ext_vector_type(16))) _Float16 v16h;
typedef __attribute__((ext_vector_type(8)))  _Float16 v8h;
typedef __attribute__((ext_vector_type(8)))  float    v8f;

#define B_   16384
#define F_   39
#define V_   100000
#define D_   64
#define H_   400
#define IN_  2496   // F*D = 39*64
#define LDH_ 448    // H padded to 7*64 (K-stage size 64)
#define NCA_ 416    // allocated columns in transposed weights (staging pads 80->96 cols)

// ---- async global->LDS copy (CDNA5 ASYNCcnt path), with synchronous fallback ----
#if defined(__has_builtin)
#if __has_builtin(__builtin_amdgcn_global_load_async_to_lds_b128) && \
    __has_builtin(__builtin_amdgcn_s_wait_asynccnt)
#define HAVE_ASYNC_LDS 1
#endif
#endif

#ifdef HAVE_ASYNC_LDS
typedef int v4i __attribute__((vector_size(16)));
typedef __attribute__((address_space(1))) v4i* gas_v4i;   // global (device) ptr
typedef __attribute__((address_space(3))) v4i* las_v4i;   // LDS (shared) ptr
#define CP128(ldsp, globp) \
  __builtin_amdgcn_global_load_async_to_lds_b128((gas_v4i)(globp), (las_v4i)(ldsp), 0, 0)
#define AWAIT(n) __builtin_amdgcn_s_wait_asynccnt(n)
#else
#define CP128(ldsp, globp) (*(uint4*)(ldsp) = *(const uint4*)(globp))
#define AWAIT(n) ((void)0)
#endif

__device__ __forceinline__ float wave_sum(float v) {
  #pragma unroll
  for (int o = 16; o > 0; o >>= 1) v += __shfl_xor(v, o, 32);
  return v;
}

// ---------------- weight transpose + f32 -> f16 convert (Wt[n*Kp + k]) ----------------
__global__ void k_prep_wt(const float* __restrict__ W, h16* __restrict__ Wt,
                          int K, int N, int Kp) {
  int i = blockIdx.x * blockDim.x + threadIdx.x;
  int total = N * Kp;
  if (i >= total) return;
  int n = i / Kp, k = i - n * Kp;
  Wt[i] = (k < K) ? (h16)W[(size_t)k * N + n] : (h16)0.0f;
}

// ---------------- embedding gather + FwFM first/second order + f16 flat ----------------
__global__ void __launch_bounds__(128)
k_gather_fwfm(const int* __restrict__ x, const float* __restrict__ embs,
              const float* __restrict__ fcov, const float* __restrict__ flin,
              h16* __restrict__ flat, float* __restrict__ base) {
  __shared__ float s_wsym[F_ * F_];
  __shared__ float s_lin[F_ * D_];
  __shared__ float s_e[4][F_ * D_];

  for (int i = threadIdx.x; i < F_ * F_; i += blockDim.x) {
    int k = i / F_, l = i - k * F_;
    s_wsym[i] = 0.5f * (fcov[k * F_ + l] + fcov[l * F_ + k]);
  }
  for (int i = threadIdx.x; i < F_ * D_; i += blockDim.x) s_lin[i] = flin[i];
  __syncthreads();

  const int lane = threadIdx.x & 31;
  const int wid  = threadIdx.x >> 5;
  const int b    = blockIdx.x * 4 + wid;

  const int* xr = x + b * F_;
  float* se = s_e[wid];
  h16* fr = flat + (size_t)b * IN_;

  float first = 0.0f, selft = 0.0f;
  for (int f = 0; f < F_; ++f) {
    int idx = xr[f];
    const float* er = embs + ((size_t)f * V_ + (size_t)idx) * D_;
    float e0 = er[lane];
    float e1 = er[lane + 32];
    se[f * D_ + lane]      = e0;
    se[f * D_ + lane + 32] = e1;
    fr[f * D_ + lane]      = (h16)e0;
    fr[f * D_ + lane + 32] = (h16)e1;
    first += e0 * s_lin[f * D_ + lane] + e1 * s_lin[f * D_ + lane + 32];
    selft += e0 * e0 + e1 * e1;
  }
  __syncthreads();

  float wsum = 0.0f;
  for (int k = 0; k < F_; ++k) {
    float t0 = 0.0f, t1 = 0.0f;
    for (int l = 0; l < F_; ++l) {
      float w = s_wsym[k * F_ + l];
      t0 += w * se[l * D_ + lane];
      t1 += w * se[l * D_ + lane + 32];
    }
    wsum += t0 * se[k * D_ + lane] + t1 * se[k * D_ + lane + 32];
  }

  first = wave_sum(first);
  selft = wave_sum(selft);
  wsum  = wave_sum(wsum);
  if (lane == 0) base[b] = first + 0.5f * (wsum - selft);
}

// ---------------- f16 WMMA GEMM + bias + ReLU:  C = relu(A[BxK] * Wt[NxK]^T + bias) ----
// Block: 256 threads (8 waves). Each wave owns one 16-row M tile; the whole block shares
// one 80-column N group. Per 64-wide K stage: the block async-copies a 96x64 B panel
// (80 used + 16 pad so transfers divide evenly) into LDS once, each wave async-copies
// its 16x64 A panel. Double-buffered: exactly 7 b128 async ops per wave per stage
// (3 B + 4 A, all lanes active) -> deterministic s_wait_asynccnt immediates.
__global__ void __launch_bounds__(256)
k_gemm_relu(const h16* __restrict__ A, const h16* __restrict__ Wt,
            const float* __restrict__ bias, h16* __restrict__ Cout,
            int K, int ldOut, int MGB) {
  __shared__ h16 s_b[2][96 * 64];       // [buf][n*64 + k]  (12 KB per buf)
  __shared__ h16 s_a[2][8][16 * 64];    // [buf][wave][m*64 + k] (2 KB per wave per buf)

  const int lane = threadIdx.x & 31;
  const int wid  = threadIdx.x >> 5;
  const int mg   = blockIdx.x % MGB;
  const int ng   = blockIdx.x / MGB;
  const int mtile = mg * 8 + wid;
  const int r0    = mtile * 16;
  const int kbA   = (lane < 16) ? 0 : 16;  // 16-bit A frag: lane halves own K 0..15 vs 16..31
  const int kbB   = (lane < 16) ? 0 : 16;  // 16-bit B frag: lane halves own K 0..15 vs 16..31
  const int m15   = lane & 15;

  const h16* wcol = Wt + (size_t)(ng * 80) * K;   // base of this block's columns
  const h16* arow = A + (size_t)r0 * K;           // base of this wave's 16 rows

  const int NS = K >> 6;                           // number of 64-wide K stages
  v8f acc[5] = {};

  // B panel: 96 cols x 64 k = 12288 B = 768 x 16B; 3 transfers per thread.
  // A panel: 16 rows x 64 k =  2048 B = 128 x 16B; 4 transfers per lane.
#define STAGE(bufi, si)                                                          \
  {                                                                              \
    const h16* wsrc = wcol + (size_t)(si) * 64;                                  \
    _Pragma("unroll")                                                            \
    for (int j = 0; j < 3; ++j) {                                                \
      int i = threadIdx.x + j * 256;                                             \
      int n = i >> 3;                                                            \
      int ko = (i & 7) * 8;                                                      \
      CP128(&s_b[bufi][n * 64 + ko], wsrc + (size_t)n * K + ko);                 \
    }                                                                            \
    const h16* asrc = arow + (size_t)(si) * 64;                                  \
    _Pragma("unroll")                                                            \
    for (int j = 0; j < 4; ++j) {                                                \
      int i = lane + j * 32;                                                     \
      int m = i >> 3;                                                            \
      int ko = (i & 7) * 8;                                                      \
      CP128(&s_a[bufi][wid][m * 64 + ko], asrc + (size_t)m * K + ko);            \
    }                                                                            \
  }

  STAGE(0, 0);

  for (int s = 0; s < NS; ++s) {
    const int buf = s & 1;
    if (s + 1 < NS) {
      STAGE(buf ^ 1, s + 1);
      AWAIT(7);                // the 7 just-issued may be outstanding; stage s is done
    } else {
      AWAIT(0);
    }
    __syncthreads();           // everyone's B (and A) panels for stage s are ready

    const h16* sa = s_a[buf][wid];
    const h16* sb = s_b[buf];
    #pragma unroll
    for (int c = 0; c < 2; ++c) {        // two 32-wide WMMA chunks per 64-wide stage
      const int kb = c * 32;
      v16h a;
      ((v8h*)&a)[0] = *(const v8h*)(sa + m15 * 64 + kb + kbA);
      ((v8h*)&a)[1] = *(const v8h*)(sa + m15 * 64 + kb + kbA + 8);
      #pragma unroll
      for (int t = 0; t < 5; ++t) {
        v16h bfr = *(const v16h*)(sb + (t * 16 + m15) * 64 + kb + kbB);
        acc[t] = __builtin_amdgcn_wmma_f32_16x16x32_f16(
            false, a, false, bfr, (short)0, acc[t], false, false);
      }
    }
    __syncthreads();           // all waves done reading before this buffer is restaged
  }
#undef STAGE

  #pragma unroll
  for (int t = 0; t < 5; ++t) {
    int ncol = (ng * 5 + t) * 16 + m15;
    float bz = bias[ncol];
    #pragma unroll
    for (int i = 0; i < 8; ++i) {
      int r = r0 + i + ((lane < 16) ? 0 : 8);   // C/D layout: VGPR i -> row i (+8 for hi half)
      float v = acc[t][i] + bz;
      v = v > 0.0f ? v : 0.0f;
      Cout[(size_t)r * ldOut + ncol] = (h16)v;
    }
  }
}

// ---------------- final GEMV (h3 . w4) + base + sigmoid ----------------
__global__ void __launch_bounds__(256)
k_head(const h16* __restrict__ h3, const float* __restrict__ w4,
       const float* __restrict__ b4, const float* __restrict__ base,
       float* __restrict__ out) {
  const int lane = threadIdx.x & 31;
  const int wid  = threadIdx.x >> 5;
  const int b    = blockIdx.x * 8 + wid;
  const h16* hr = h3 + (size_t)b * LDH_;
  float acc = 0.0f;
  for (int k = lane; k < H_; k += 32) acc += (float)hr[k] * w4[k];
  acc = wave_sum(acc);
  if (lane == 0) {
    float z = base[b] + acc + b4[0];
    out[b] = 1.0f / (1.0f + __expf(-z));
  }
}

extern "C" void kernel_launch(void* const* d_in, const int* in_sizes, int n_in,
                              void* d_out, int out_size, void* d_ws, size_t ws_size,
                              hipStream_t stream) {
  (void)in_sizes; (void)n_in; (void)out_size; (void)ws_size;
  const int*   x    = (const int*)  d_in[0];
  const float* embs = (const float*)d_in[1];
  const float* fcov = (const float*)d_in[2];
  const float* flin = (const float*)d_in[3];
  const float* w1   = (const float*)d_in[4];
  const float* b1   = (const float*)d_in[5];
  const float* w2   = (const float*)d_in[6];
  const float* b2   = (const float*)d_in[7];
  const float* w3   = (const float*)d_in[8];
  const float* b3   = (const float*)d_in[9];
  const float* w4   = (const float*)d_in[10];
  const float* b4   = (const float*)d_in[11];
  float* out = (float*)d_out;

  char* ws = (char*)d_ws;
  size_t off = 0;
  auto alloc = [&](size_t bytes) -> char* {
    char* p = ws + off;
    off = (off + bytes + 255) & ~(size_t)255;
    return p;
  };
  h16*   flat = (h16*)  alloc((size_t)B_ * IN_ * sizeof(h16));
  h16*   h1   = (h16*)  alloc((size_t)B_ * LDH_ * sizeof(h16));
  h16*   h2   = (h16*)  alloc((size_t)B_ * LDH_ * sizeof(h16));
  h16*   h3   = (h16*)  alloc((size_t)B_ * LDH_ * sizeof(h16));
  h16*   Wt1  = (h16*)  alloc((size_t)NCA_ * IN_ * sizeof(h16));   // 16 pad cols for staging
  h16*   Wt2  = (h16*)  alloc((size_t)NCA_ * LDH_ * sizeof(h16));
  h16*   Wt3  = (h16*)  alloc((size_t)NCA_ * LDH_ * sizeof(h16));
  float* base = (float*)alloc((size_t)B_ * sizeof(float));

  // weight prep (tiny); staging may read (but never consumes) cols 400..415
  {
    int total = H_ * IN_;
    k_prep_wt<<<(total + 255) / 256, 256, 0, stream>>>(w1, Wt1, IN_, H_, IN_);
  }
  {
    int total = H_ * LDH_;
    k_prep_wt<<<(total + 255) / 256, 256, 0, stream>>>(w2, Wt2, H_, H_, LDH_);
    k_prep_wt<<<(total + 255) / 256, 256, 0, stream>>>(w3, Wt3, H_, H_, LDH_);
  }

  // gather + FwFM terms + f16 flat
  k_gather_fwfm<<<B_ / 4, 128, 0, stream>>>(x, embs, fcov, flin, flat, base);

  // MLP: three WMMA GEMMs with fused bias+ReLU (async-LDS double-buffered)
  const int MGB = (B_ / 16) / 8;  // 128 M-tile groups, 8 waves per block
  k_gemm_relu<<<MGB * 5, 256, 0, stream>>>(flat, Wt1, b1, h1, IN_,  LDH_, MGB);
  k_gemm_relu<<<MGB * 5, 256, 0, stream>>>(h1,   Wt2, b2, h2, LDH_, LDH_, MGB);
  k_gemm_relu<<<MGB * 5, 256, 0, stream>>>(h2,   Wt3, b3, h3, LDH_, LDH_, MGB);

  // head: dot(h3, w4) + base -> sigmoid
  k_head<<<B_ / 8, 256, 0, stream>>>(h3, w4, b4, base, out);
}
